// LiquidRNN_87110526697522
// MI455X (gfx1250) — compile-verified, matrix operand order checked by reading, not run on hardware
//
#include <hip/hip_runtime.h>
#include <cmath>

// ---------------------------------------------------------------------------
// LiquidRNN for MI455X (gfx1250): bf16 WMMA GEMMs, LDS-resident RNN scan,
// persistent grid-barrier liquid scan.
// ---------------------------------------------------------------------------

#define S_DIM 1024
#define N_DIM 128
#define D_DIM 512
#define H_DIM 512
#define DT_CONST 0.1f
#define LIQ_BLOCKS 64

typedef __attribute__((ext_vector_type(16))) __bf16 v16bf;
typedef __attribute__((ext_vector_type(8)))  float  v8f;
typedef __attribute__((ext_vector_type(4)))  unsigned int v4u;

union FragBF {
  v16bf v;
  __bf16 e[16];
  v4u    q[2];
};

static __device__ __forceinline__ v8f wmma_bf16(const FragBF& a, const FragBF& b, v8f c) {
  // D = A(16x32 bf16) * B(32x16 bf16) + C(16x16 f32)
  return __builtin_amdgcn_wmma_f32_16x16x32_bf16(false, a.v, false, b.v, (short)0, c,
                                                 false, false);
}

// Load a 16x32 bf16 fragment from row-major storage (row stride = ld elements).
// ISA layout: lane<16 holds row=lane, K={0..7,16..23}; lane>=16 holds K={8..15,24..31}.
static __device__ __forceinline__ void load_frag(FragBF& f, const __bf16* base, int ld,
                                                 int lane, int kc) {
  const int lrow = lane & 15;
  const int koff = ((lane >> 4) << 3) + kc;
  const __bf16* p = base + (size_t)lrow * ld + koff;
  f.q[0] = *(const v4u*)p;
  f.q[1] = *(const v4u*)(p + 16);
}

// ---------------------------------------------------------------------------
// Kernel 0: convert weights fp32 -> bf16, precompute DT/clip(tau), init barrier
// ---------------------------------------------------------------------------
__global__ void prep_kernel(const float* __restrict__ W_ih, const float* __restrict__ W_hh,
                            const float* __restrict__ fc_in_w,
                            const float* __restrict__ fc_rec_w,
                            const float* __restrict__ tau,
                            __bf16* __restrict__ wih, __bf16* __restrict__ whh,
                            __bf16* __restrict__ wfin, __bf16* __restrict__ wfrec,
                            float* __restrict__ dtau, unsigned* __restrict__ bar) {
  size_t i = (size_t)blockIdx.x * blockDim.x + threadIdx.x;
  size_t stride = (size_t)gridDim.x * blockDim.x;
  for (size_t k = i; k < (size_t)H_DIM * D_DIM; k += stride) {
    wih[k]  = (__bf16)W_ih[k];
    whh[k]  = (__bf16)W_hh[k];
    wfin[k] = (__bf16)fc_in_w[k];
    wfrec[k] = (__bf16)fc_rec_w[k];
  }
  if (i < H_DIM) {
    float tc = fminf(fmaxf(tau[i], 0.1f), 5.0f);
    dtau[i] = DT_CONST / tc;
  }
  if (i < 2) bar[i] = 0u;
}

// ---------------------------------------------------------------------------
// Kernel 1: xi = bf16( x @ W_ih^T + b_ih )   (131072 x 512) x (512 x 512)
// One 16x16 output tile per wave, K-loop of 16 WMMAs; A converted fp32->bf16
// in registers.
// ---------------------------------------------------------------------------
__global__ __launch_bounds__(256) void gemm_xi_kernel(const float* __restrict__ x,
                                                      const __bf16* __restrict__ Wih,
                                                      const float* __restrict__ b_ih,
                                                      __bf16* __restrict__ xi) {
  const int lane = threadIdx.x & 31;
  const int wave = threadIdx.x >> 5;
  const long id = (long)blockIdx.x * 8 + wave;       // tile id, 8192*32 tiles
  const int tn = (int)(id & 31);
  const long m0 = (id >> 5) * 16;
  const int n0 = tn * 16;
  const int lrow = lane & 15;
  const int koff = (lane >> 4) << 3;

  const float* arow = x + (size_t)(m0 + lrow) * D_DIM;
  const __bf16* brow = Wih + (size_t)(n0 + lrow) * D_DIM;

  v8f c;
#pragma unroll
  for (int r = 0; r < 8; ++r) c[r] = 0.0f;

  for (int kc = 0; kc < D_DIM; kc += 32) {
    FragBF a, b;
    const __bf16* bp = brow + kc + koff;
    b.q[0] = *(const v4u*)bp;
    b.q[1] = *(const v4u*)(bp + 16);
    const float* ap = arow + kc + koff;
    __builtin_prefetch(ap + 32, 0, 0);
#pragma unroll
    for (int j = 0; j < 8; ++j) {
      a.e[j]     = (__bf16)ap[j];
      a.e[8 + j] = (__bf16)ap[16 + j];
    }
    c = wmma_bf16(a, b, c);
  }

  const int col = n0 + lrow;
  const float bias = b_ih[col];
#pragma unroll
  for (int r = 0; r < 8; ++r) {
    long m = m0 + r + ((lane >> 4) << 3);
    xi[(size_t)m * H_DIM + col] = (__bf16)(c[r] + bias);
  }
}

// ---------------------------------------------------------------------------
// Kernel 2: RNN scan, persistent single workgroup (32 waves, one WGP).
// h (128x512 bf16) double-buffered in LDS (2 x 128KB <= 320KB WGP LDS).
// Wave w owns output columns [16w, 16w+16) and all 8 row tiles; per step:
// 128 WMMAs/wave + tanh epilogue + one workgroup barrier.
// ---------------------------------------------------------------------------
__global__ __launch_bounds__(1024) void rnn_scan_kernel(const __bf16* __restrict__ xi,
                                                        const __bf16* __restrict__ Whh,
                                                        const float* __restrict__ b_hh,
                                                        __bf16* __restrict__ ys) {
  extern __shared__ __bf16 hsh[];               // 2 * 128 * 512 bf16
  __bf16* const hbuf0 = hsh;
  __bf16* const hbuf1 = hsh + (size_t)N_DIM * H_DIM;

  const int tid = threadIdx.x;
  const int wave = tid >> 5;
  const int lane = tid & 31;

  for (int i = tid; i < N_DIM * H_DIM; i += 1024) hbuf0[i] = (__bf16)0.0f;
  __syncthreads();

  const int n0 = wave * 16;                     // this wave's column block
  const int lrow = lane & 15;
  const int koff = (lane >> 4) << 3;
  const int col = n0 + lrow;
  const float bias = b_hh[col];
  const __bf16* const brow = Whh + (size_t)(n0 + lrow) * H_DIM;

#pragma unroll 1
  for (int t = 0; t < S_DIM; ++t) {
    const __bf16* hin = (t & 1) ? hbuf1 : hbuf0;
    __bf16* hout      = (t & 1) ? hbuf0 : hbuf1;
    const __bf16* xit = xi + (size_t)t * N_DIM * H_DIM;
    __bf16* yst       = ys + (size_t)t * N_DIM * H_DIM;

    v8f c[8];
#pragma unroll
    for (int i = 0; i < 8; ++i)
#pragma unroll
      for (int r = 0; r < 8; ++r) c[i][r] = 0.0f;

    for (int kc = 0; kc < H_DIM; kc += 32) {
      FragBF b;
      const __bf16* bp = brow + kc + koff;
      b.q[0] = *(const v4u*)bp;
      b.q[1] = *(const v4u*)(bp + 16);
#pragma unroll
      for (int i = 0; i < 8; ++i) {
        FragBF a;
        const __bf16* ap = hin + (size_t)(i * 16 + lrow) * H_DIM + kc + koff;
        a.q[0] = *(const v4u*)ap;
        a.q[1] = *(const v4u*)(ap + 16);
        c[i] = wmma_bf16(a, b, c[i]);
      }
    }

#pragma unroll
    for (int i = 0; i < 8; ++i) {
#pragma unroll
      for (int r = 0; r < 8; ++r) {
        const int m = i * 16 + r + ((lane >> 4) << 3);
        const size_t idx = (size_t)m * H_DIM + col;
        float v = c[i][r] + (float)xit[idx] + bias;
        __bf16 hb = (__bf16)tanhf(v);
        hout[idx] = hb;
        yst[idx] = hb;
      }
    }
    __syncthreads();   // double-buffered h: one barrier per step
  }
}

// ---------------------------------------------------------------------------
// Kernel 3: xin[n][s][:] = bf16( ys[s][n][:] @ fc_in_w^T + fc_in_b )
// stored (N, S, H) so the liquid scan reads a contiguous (S,H) slab per step.
// ---------------------------------------------------------------------------
__global__ __launch_bounds__(256) void gemm_xin_kernel(const __bf16* __restrict__ ys,
                                                       const __bf16* __restrict__ Wfin,
                                                       const float* __restrict__ b,
                                                       __bf16* __restrict__ xin) {
  const int lane = threadIdx.x & 31;
  const int wave = threadIdx.x >> 5;
  const long id = (long)blockIdx.x * 8 + wave;  // 128 * 64 * 32 tiles
  const int n = (int)(id >> 11);
  const int rem = (int)(id & 2047);
  const int tm = rem >> 5;
  const int tn = rem & 31;
  const int n0 = tn * 16;
  const int lrow = lane & 15;
  const int koff = (lane >> 4) << 3;

  const __bf16* arow = ys + ((size_t)(tm * 16 + lrow) * N_DIM + n) * H_DIM;
  const __bf16* brow = Wfin + (size_t)(n0 + lrow) * H_DIM;

  v8f c;
#pragma unroll
  for (int r = 0; r < 8; ++r) c[r] = 0.0f;

  for (int kc = 0; kc < H_DIM; kc += 32) {
    FragBF a, bb;
    const __bf16* ap = arow + kc + koff;
    a.q[0] = *(const v4u*)ap;
    a.q[1] = *(const v4u*)(ap + 16);
    const __bf16* bp = brow + kc + koff;
    bb.q[0] = *(const v4u*)bp;
    bb.q[1] = *(const v4u*)(bp + 16);
    c = wmma_bf16(a, bb, c);
  }

  const int col = n0 + lrow;
  const float bias = b[col];
#pragma unroll
  for (int r = 0; r < 8; ++r) {
    const int s = tm * 16 + r + ((lane >> 4) << 3);
    xin[((size_t)n * S_DIM + s) * H_DIM + col] = (__bf16)(c[r] + bias);
  }
}

// ---------------------------------------------------------------------------
// Grid-wide sense-reversing barrier (all LIQ_BLOCKS workgroups co-resident).
// ---------------------------------------------------------------------------
static __device__ __forceinline__ void grid_barrier(unsigned* bar, unsigned nb) {
  __threadfence();      // release this thread's global stores to device scope
  __syncthreads();
  if (threadIdx.x == 0) {
    volatile unsigned* vbar = (volatile unsigned*)bar;
    unsigned gen = vbar[1];
    unsigned prev = atomicAdd(&bar[0], 1u);
    if (prev == nb - 1u) {
      vbar[0] = 0u;
      __threadfence();
      atomicAdd(&bar[1], 1u);
    } else {
      while (vbar[1] == gen) __builtin_amdgcn_s_sleep(2);
    }
  }
  __syncthreads();
  __threadfence();
}

// ---------------------------------------------------------------------------
// Kernel 4: liquid scan, persistent 64 workgroups x 8 waves. Each wave owns
// column tile tn=gw&31 and 4 row tiles {gw>>5 + 16i}; h double-buffered in
// global ws (fp32 state + bf16 WMMA operand copy). One grid barrier per step.
// ---------------------------------------------------------------------------
__global__ __launch_bounds__(256) void liquid_scan_kernel(const __bf16* __restrict__ xin,
                                                          const __bf16* __restrict__ Wrec,
                                                          const float* __restrict__ b_rec,
                                                          const float* __restrict__ dtau,
                                                          float* __restrict__ hF,
                                                          __bf16* __restrict__ hB,
                                                          unsigned* __restrict__ bar) {
  const int tid = threadIdx.x;
  const int wave = tid >> 5;
  const int lane = tid & 31;
  const int gw = blockIdx.x * 8 + wave;          // 0..511
  const int tn = gw & 31;
  const int tmb = gw >> 5;                       // 0..15
  const int n0 = tn * 16;
  const int lrow = lane & 15;
  const int koff = (lane >> 4) << 3;
  const int col = n0 + lrow;
  const float bias = b_rec[col];
  const float dt = dtau[col];
  const __bf16* const brow = Wrec + (size_t)(n0 + lrow) * H_DIM;

  // zero h buffer 0 (initial liquid state)
  for (size_t i = (size_t)blockIdx.x * 256 + tid; i < (size_t)S_DIM * H_DIM;
       i += (size_t)LIQ_BLOCKS * 256) {
    hF[i] = 0.0f;
    hB[i] = (__bf16)0.0f;
  }
  grid_barrier(bar, LIQ_BLOCKS);

#pragma unroll 1
  for (int t = 0; t < N_DIM; ++t) {
    const int p = t & 1;
    const float* hFin = hF + (size_t)p * S_DIM * H_DIM;
    float* hFout = hF + (size_t)(1 - p) * S_DIM * H_DIM;
    const __bf16* hBin = hB + (size_t)p * S_DIM * H_DIM;
    __bf16* hBout = hB + (size_t)(1 - p) * S_DIM * H_DIM;
    const __bf16* xt = xin + (size_t)t * S_DIM * H_DIM;

    v8f c[4];
#pragma unroll
    for (int i = 0; i < 4; ++i)
#pragma unroll
      for (int r = 0; r < 8; ++r) c[i][r] = 0.0f;

    for (int kc = 0; kc < H_DIM; kc += 32) {
      FragBF b;
      const __bf16* bp = brow + kc + koff;
      b.q[0] = *(const v4u*)bp;
      b.q[1] = *(const v4u*)(bp + 16);
#pragma unroll
      for (int i = 0; i < 4; ++i) {
        FragBF a;
        const int tm = tmb + 16 * i;
        const __bf16* ap = hBin + (size_t)(tm * 16 + lrow) * H_DIM + kc + koff;
        a.q[0] = *(const v4u*)ap;
        a.q[1] = *(const v4u*)(ap + 16);
        c[i] = wmma_bf16(a, b, c[i]);
      }
    }

#pragma unroll
    for (int i = 0; i < 4; ++i) {
      const int tm = tmb + 16 * i;
#pragma unroll
      for (int r = 0; r < 8; ++r) {
        const int s = tm * 16 + r + ((lane >> 4) << 3);
        const size_t idx = (size_t)s * H_DIM + col;
        float act = tanhf(c[i][r] + (float)xt[idx] + bias);
        float h = hFin[idx];                      // owned by this thread every step
        float hn = h + dt * (act - h);
        hFout[idx] = hn;
        hBout[idx] = (__bf16)hn;
      }
    }
    grid_barrier(bar, LIQ_BLOCKS);
  }
}

// ---------------------------------------------------------------------------
// Kernel 5: copy final liquid state (buffer 0 after 128 steps) to d_out.
// ---------------------------------------------------------------------------
__global__ void copy_out_kernel(const float* __restrict__ src, float* __restrict__ dst,
                                int n) {
  int i = blockIdx.x * blockDim.x + threadIdx.x;
  const int stride = gridDim.x * blockDim.x;
  for (; i < n; i += stride) dst[i] = src[i];
}

// ---------------------------------------------------------------------------
extern "C" void kernel_launch(void* const* d_in, const int* in_sizes, int n_in,
                              void* d_out, int out_size, void* d_ws, size_t ws_size,
                              hipStream_t stream) {
  (void)in_sizes; (void)n_in; (void)out_size; (void)ws_size;
  const float* x        = (const float*)d_in[0];
  const float* W_ih     = (const float*)d_in[1];
  const float* W_hh     = (const float*)d_in[2];
  const float* b_ih     = (const float*)d_in[3];
  const float* b_hh     = (const float*)d_in[4];
  const float* fc_in_w  = (const float*)d_in[5];
  const float* fc_in_b  = (const float*)d_in[6];
  const float* fc_rec_w = (const float*)d_in[7];
  const float* fc_rec_b = (const float*)d_in[8];
  const float* tau      = (const float*)d_in[9];

  char* ws = (char*)d_ws;
  size_t off = 0;
  auto take = [&](size_t bytes) -> char* {
    char* p = ws + off;
    off += (bytes + 255) & ~(size_t)255;
    return p;
  };
  const size_t wmat = (size_t)H_DIM * D_DIM * sizeof(__bf16);
  const size_t act3 = (size_t)S_DIM * N_DIM * H_DIM * sizeof(__bf16);

  __bf16* wih_b   = (__bf16*)take(wmat);
  __bf16* whh_b   = (__bf16*)take(wmat);
  __bf16* wfin_b  = (__bf16*)take(wmat);
  __bf16* wfrec_b = (__bf16*)take(wmat);
  float*  dtau    = (float*)take(H_DIM * sizeof(float));
  unsigned* bar   = (unsigned*)take(256);
  __bf16* xi      = (__bf16*)take(act3);
  __bf16* ys      = (__bf16*)take(act3);
  __bf16* xin     = (__bf16*)take(act3);
  float*  hF      = (float*)take((size_t)2 * S_DIM * H_DIM * sizeof(float));
  __bf16* hB      = (__bf16*)take((size_t)2 * S_DIM * H_DIM * sizeof(__bf16));

  prep_kernel<<<512, 256, 0, stream>>>(W_ih, W_hh, fc_in_w, fc_rec_w, tau,
                                       wih_b, whh_b, wfin_b, wfrec_b, dtau, bar);

  // xi = x @ W_ih^T + b_ih : 262144 tiles / 8 waves per block
  gemm_xi_kernel<<<32768, 256, 0, stream>>>(x, wih_b, b_ih, xi);

  // RNN scan: 1 workgroup, 32 waves, 256KB dynamic LDS (double-buffered h)
  const size_t rnn_lds = (size_t)2 * N_DIM * H_DIM * sizeof(__bf16);
  hipFuncSetAttribute(reinterpret_cast<const void*>(rnn_scan_kernel),
                      hipFuncAttributeMaxDynamicSharedMemorySize, (int)rnn_lds);
  rnn_scan_kernel<<<1, 1024, rnn_lds, stream>>>(xi, whh_b, b_hh, ys);

  // xin = ys @ fc_in_w^T + fc_in_b, stored (N, S, H)
  gemm_xin_kernel<<<32768, 256, 0, stream>>>(ys, wfin_b, fc_in_b, xin);

  // liquid scan: persistent 64 WGs with grid barrier, 128 steps
  liquid_scan_kernel<<<LIQ_BLOCKS, 256, 0, stream>>>(xin, wfrec_b, fc_rec_b, dtau,
                                                     hF, hB, bar);

  // final state is in hF buffer 0 after an even number of steps
  copy_out_kernel<<<512, 256, 0, stream>>>(hF, (float*)d_out, S_DIM * H_DIM);
}
